// LlamaMoE_48361331752997
// MI455X (gfx1250) — matrix operand order repaired and checked by reading, not compiled, not run on hardware
//
#include <hip/hip_runtime.h>
#include <hip/hip_bf16.h>
#include <math.h>

// ---------------------------------------------------------------------------
// LlamaMoE for MI455X (gfx1250): bf16 WMMA GEMMs, top-2 gathered dispatch.
// H=2048, I=4096, E=8, K=2, N=2048 tokens. f32 in / f32 out.
// Activation tiles staged to LDS via global_load_async_to_lds_b128 (ASYNCcnt),
// weight tiles streamed f32 -> converted in-register -> bf16 LDS (transposed).
// ---------------------------------------------------------------------------

#define H_DIM 2048
#define I_DIM 4096
#define E_NUM 8
#define NTOK  2048

#define MT 64    // M tile (rows) per block
#define CT 128   // output-col tile per block
#define KT 32    // K step (matches 16x16x32 bf16 WMMA)
#define XS_STRIDE 40   // 32 + 8 pad (bf16 elems); 80B rows, 16B aligned
#define WS_STRIDE 40

typedef __attribute__((ext_vector_type(16))) __bf16 bf16x16;
typedef __attribute__((ext_vector_type(8)))  float  f32x8;

__device__ __forceinline__ unsigned short f32_to_bf16(float f) {
    union { float f; unsigned u; } v; v.f = f;
    unsigned u = v.u;
    unsigned r = u + 0x7fffu + ((u >> 16) & 1u);   // round-to-nearest-even
    return (unsigned short)(r >> 16);
}

__device__ __forceinline__ float silu(float g) {
    return g / (1.0f + expf(-g));
}

// Async copy 16 bytes global -> LDS (per lane). Tracked by ASYNCcnt.
__device__ __forceinline__ void async_copy_b128(const void* gsrc, void* lds_dst) {
    unsigned ldsoff = (unsigned)(size_t)lds_dst;   // low 32 bits = LDS offset
    asm volatile("global_load_async_to_lds_b128 %0, %1, off"
                 :: "v"(ldsoff), "v"(gsrc) : "memory");
}
__device__ __forceinline__ void async_wait0() {
    asm volatile("s_wait_asynccnt 0x0" ::: "memory");
}

// ---------------------------------------------------------------------------
__global__ void init_counts_kernel(int* counts) {
    int i = threadIdx.x;
    if (i < E_NUM) counts[i] = 0;
    if (i == E_NUM) counts[E_NUM] = NTOK;   // slot E holds "dense" count
}

// x f32 -> bf16, 4 elems/thread
__global__ void cvt_x_kernel(const float* __restrict__ x,
                             unsigned short* __restrict__ xb, int n4) {
    int i = blockIdx.x * blockDim.x + threadIdx.x;
    if (i >= n4) return;
    float4 f = ((const float4*)x)[i];
    ushort4 o;
    o.x = f32_to_bf16(f.x); o.y = f32_to_bf16(f.y);
    o.z = f32_to_bf16(f.z); o.w = f32_to_bf16(f.w);
    ((ushort4*)xb)[i] = o;
}

// ---------------------------------------------------------------------------
// Router: one wave per token. logits = x @ gate_w, softmax fp32, top-2,
// renormalize; writes comb[N,E] and per-expert token lists.
__global__ void router_kernel(const float* __restrict__ x,
                              const float* __restrict__ gw,
                              float* __restrict__ comb,
                              int* __restrict__ counts,
                              int* __restrict__ lists) {
    int wave = threadIdx.x >> 5;
    int lane = threadIdx.x & 31;
    int token = blockIdx.x * 8 + wave;
    if (token >= NTOK) return;

    const float* xr = x + (size_t)token * H_DIM;
    float acc[E_NUM];
#pragma unroll
    for (int e = 0; e < E_NUM; ++e) acc[e] = 0.0f;

    for (int h = lane; h < H_DIM; h += 32) {
        float xv = xr[h];
        const float* g = gw + (size_t)h * E_NUM;
#pragma unroll
        for (int e = 0; e < E_NUM; ++e) acc[e] += xv * g[e];
    }
#pragma unroll
    for (int e = 0; e < E_NUM; ++e) {
        float v = acc[e];
        for (int off = 16; off > 0; off >>= 1) v += __shfl_xor(v, off, 32);
        acc[e] = v;
    }
    if (lane == 0) {
        float mx = acc[0];
#pragma unroll
        for (int e = 1; e < E_NUM; ++e) mx = fmaxf(mx, acc[e]);
        float p[E_NUM]; float s = 0.0f;
#pragma unroll
        for (int e = 0; e < E_NUM; ++e) { p[e] = expf(acc[e] - mx); s += p[e]; }
        float inv = 1.0f / s;
#pragma unroll
        for (int e = 0; e < E_NUM; ++e) p[e] *= inv;
        // top-2, ties -> lowest index (matches jax top_k)
        int i0 = 0;
#pragma unroll
        for (int e = 1; e < E_NUM; ++e) if (p[e] > p[i0]) i0 = e;
        int i1 = (i0 == 0) ? 1 : 0;
#pragma unroll
        for (int e = 0; e < E_NUM; ++e) if (e != i0 && p[e] > p[i1]) i1 = e;
        float w0 = p[i0], w1 = p[i1], sw = w0 + w1;
        w0 /= sw; w1 /= sw;
#pragma unroll
        for (int e = 0; e < E_NUM; ++e) comb[token * E_NUM + e] = 0.0f;
        comb[token * E_NUM + i0] = w0;
        comb[token * E_NUM + i1] = w1;
        int p0 = atomicAdd(&counts[i0], 1); lists[i0 * NTOK + p0] = token;
        int p1 = atomicAdd(&counts[i1], 1); lists[i1 * NTOK + p1] = token;
    }
}

// ---------------------------------------------------------------------------
// Gate/Up + SiLU-mul GEMM: T[r, c] = scale(r) * silu(X@Wg)[r,c] * (X@Wu)[r,c]
// X rows gathered through `tokens` (nullptr -> identity). Output bf16.
__global__ __launch_bounds__(256)
void gateup_kernel(const unsigned short* __restrict__ xb,   // [NTOK,H] bf16
                   const float* __restrict__ wgu,           // [H, 2I] f32
                   unsigned short* __restrict__ tbuf,       // [*, I] bf16
                   const int* __restrict__ tokens,          // nullable
                   const float* __restrict__ rowscale,      // nullable: comb+e
                   const int* __restrict__ countp) {
    __shared__ __align__(16) unsigned short Xs[MT * XS_STRIDE];
    __shared__ __align__(16) unsigned short Gs[CT * WS_STRIDE];
    __shared__ __align__(16) unsigned short Us[CT * WS_STRIDE];
    __shared__ float scale_s[MT];
    __shared__ int   tok_s[MT];

    const int count = *countp;
    const int m0 = blockIdx.y * MT;
    if (m0 >= count) return;
    const int c0 = blockIdx.x * CT;
    const int tid = threadIdx.x;

    if (tid < MT) {
        int row = m0 + tid;
        bool valid = row < count;
        int token = valid ? (tokens ? tokens[row] : row) : 0;
        float sc = valid ? (rowscale ? rowscale[(size_t)token * E_NUM] : 1.0f) : 0.0f;
        tok_s[tid] = token;
        scale_s[tid] = sc;
    }
    __syncthreads();

    const int lane = tid & 31;
    const int wv = tid >> 5;
    const int wm = (wv >> 1) * 16;   // 0,16,32,48
    const int wn = (wv & 1) * 64;    // 0 or 64

    const f32x8 zero8 = {0.f,0.f,0.f,0.f,0.f,0.f,0.f,0.f};
    f32x8 accG[4], accU[4];
#pragma unroll
    for (int i = 0; i < 4; ++i) { accG[i] = zero8; accU[i] = zero8; }

    const int xrow = tid >> 2;          // 0..63
    const int xcol = (tid & 3) * 8;     // 0,8,16,24
    const int wc   = tid & 127;         // weight column within tile
    const int wk0  = (tid >> 7) * 16;   // k half: 0 or 16
    const int xtok = tok_s[xrow];
    const unsigned short* xsrc = xb + (size_t)xtok * H_DIM + xcol;
    unsigned short* xdst = &Xs[xrow * XS_STRIDE + xcol];

    for (int kt = 0; kt < H_DIM; kt += KT) {
        __syncthreads();
        // X tile -> LDS via async DMA (bf16, no conversion needed)
        async_copy_b128(xsrc + kt, xdst);
        // W gate/up tiles: coalesced f32 loads, cvt->bf16, store transposed
        {
            unsigned short tg[16], tu[16];
#pragma unroll
            for (int kk = 0; kk < 16; ++kk) {
                size_t off = (size_t)(kt + wk0 + kk) * (2 * I_DIM) + (c0 + wc);
                tg[kk] = f32_to_bf16(wgu[off]);
                tu[kk] = f32_to_bf16(wgu[off + I_DIM]);
            }
            *(uint4*)&Gs[wc * WS_STRIDE + wk0]     = *(uint4*)&tg[0];
            *(uint4*)&Gs[wc * WS_STRIDE + wk0 + 8] = *(uint4*)&tg[8];
            *(uint4*)&Us[wc * WS_STRIDE + wk0]     = *(uint4*)&tu[0];
            *(uint4*)&Us[wc * WS_STRIDE + wk0 + 8] = *(uint4*)&tu[8];
        }
        if (kt + KT < H_DIM) {  // gfx1250 global_prefetch_b8 on next tile
            __builtin_prefetch(&wgu[(size_t)(kt + KT + wk0) * (2 * I_DIM) + (c0 + wc)], 0, 1);
        }
        async_wait0();          // async LDS fill visible before barrier
        __syncthreads();

        // A fragment: contiguous k order, same as B (lanes<16: K 0-15; else K 16-31)
        bf16x16 a;
        {
            int m  = wm + (lane & 15);
            int kb = (lane >= 16) ? 16 : 0;
            union { uint4 q[2]; bf16x16 v; } ua;
            ua.q[0] = *(const uint4*)&Xs[m * XS_STRIDE + kb];
            ua.q[1] = *(const uint4*)&Xs[m * XS_STRIDE + kb + 8];
            a = ua.v;
        }
        const int kb2 = (lane >= 16) ? 16 : 0;
#pragma unroll
        for (int c = 0; c < 4; ++c) {
            int col = wn + c * 16 + (lane & 15);
            union { uint4 q[2]; bf16x16 v; } bg, bu;
            bg.q[0] = *(const uint4*)&Gs[col * WS_STRIDE + kb2];
            bg.q[1] = *(const uint4*)&Gs[col * WS_STRIDE + kb2 + 8];
            accG[c] = __builtin_amdgcn_wmma_f32_16x16x32_bf16(
                false, a, false, bg.v, (short)0, accG[c], false, false);
            bu.q[0] = *(const uint4*)&Us[col * WS_STRIDE + kb2];
            bu.q[1] = *(const uint4*)&Us[col * WS_STRIDE + kb2 + 8];
            accU[c] = __builtin_amdgcn_wmma_f32_16x16x32_bf16(
                false, a, false, bu.v, (short)0, accU[c], false, false);
        }
    }

    // epilogue: SiLU * up * rowscale -> bf16 T
    const int ncol = lane & 15;
    const int mofs = (lane >= 16) ? 8 : 0;
#pragma unroll
    for (int c = 0; c < 4; ++c) {
#pragma unroll
        for (int r = 0; r < 8; ++r) {
            int ml = wm + r + mofs;
            int row = m0 + ml;
            if (row < count) {
                float t = silu(accG[c][r]) * accU[c][r] * scale_s[ml];
                tbuf[(size_t)row * I_DIM + (c0 + wn + c * 16 + ncol)] = f32_to_bf16(t);
            }
        }
    }
}

// ---------------------------------------------------------------------------
// Down-proj GEMM: out[token, c] (+)= (T @ Wd)[r, c], rows scattered via tokens.
__global__ __launch_bounds__(256)
void down_kernel(const unsigned short* __restrict__ tbuf,   // [*, I] bf16
                 const float* __restrict__ wd,               // [I, H] f32
                 float* __restrict__ out,                    // [NTOK, H] f32
                 const int* __restrict__ tokens,             // nullable
                 const int* __restrict__ countp,
                 int accumulate) {
    __shared__ __align__(16) unsigned short Xs[MT * XS_STRIDE];
    __shared__ __align__(16) unsigned short Ws[CT * WS_STRIDE];
    __shared__ int tok_s[MT];

    const int count = *countp;
    const int m0 = blockIdx.y * MT;
    if (m0 >= count) return;
    const int c0 = blockIdx.x * CT;
    const int tid = threadIdx.x;

    if (tid < MT) {
        int row = m0 + tid;
        tok_s[tid] = (row < count) ? (tokens ? tokens[row] : row) : 0;
    }
    __syncthreads();

    const int lane = tid & 31;
    const int wv = tid >> 5;
    const int wm = (wv >> 1) * 16;
    const int wn = (wv & 1) * 64;

    const f32x8 zero8 = {0.f,0.f,0.f,0.f,0.f,0.f,0.f,0.f};
    f32x8 acc[4];
#pragma unroll
    for (int i = 0; i < 4; ++i) acc[i] = zero8;

    const int xrow = tid >> 2;
    const int xcol = (tid & 3) * 8;
    const int wc   = tid & 127;
    const int wk0  = (tid >> 7) * 16;
    const unsigned short* tsrc = tbuf + (size_t)(m0 + xrow) * I_DIM + xcol;
    unsigned short* tdst = &Xs[xrow * XS_STRIDE + xcol];

    for (int kt = 0; kt < I_DIM; kt += KT) {
        __syncthreads();
        // T tile -> LDS via async DMA (bf16)
        async_copy_b128(tsrc + kt, tdst);
        {
            unsigned short tw[16];
#pragma unroll
            for (int kk = 0; kk < 16; ++kk) {
                tw[kk] = f32_to_bf16(wd[(size_t)(kt + wk0 + kk) * H_DIM + (c0 + wc)]);
            }
            *(uint4*)&Ws[wc * WS_STRIDE + wk0]     = *(uint4*)&tw[0];
            *(uint4*)&Ws[wc * WS_STRIDE + wk0 + 8] = *(uint4*)&tw[8];
        }
        if (kt + KT < I_DIM) {
            __builtin_prefetch(&wd[(size_t)(kt + KT + wk0) * H_DIM + (c0 + wc)], 0, 1);
        }
        async_wait0();
        __syncthreads();

        bf16x16 a;
        {
            int m  = wm + (lane & 15);
            int kb = (lane >= 16) ? 16 : 0;
            union { uint4 q[2]; bf16x16 v; } ua;
            ua.q[0] = *(const uint4*)&Xs[m * XS_STRIDE + kb];
            ua.q[1] = *(const uint4*)&Xs[m * XS_STRIDE + kb + 8];
            a = ua.v;
        }
        const int kb2 = (lane >= 16) ? 16 : 0;
#pragma unroll
        for (int c = 0; c < 4; ++c) {
            int col = wn + c * 16 + (lane & 15);
            union { uint4 q[2]; bf16x16 v; } bw;
            bw.q[0] = *(const uint4*)&Ws[col * WS_STRIDE + kb2];
            bw.q[1] = *(const uint4*)&Ws[col * WS_STRIDE + kb2 + 8];
            acc[c] = __builtin_amdgcn_wmma_f32_16x16x32_bf16(
                false, a, false, bw.v, (short)0, acc[c], false, false);
        }
    }

    const int ncol = lane & 15;
    const int mofs = (lane >= 16) ? 8 : 0;
#pragma unroll
    for (int c = 0; c < 4; ++c) {
#pragma unroll
        for (int r = 0; r < 8; ++r) {
            int ml = wm + r + mofs;
            int row = m0 + ml;
            if (row < count) {
                size_t idx = (size_t)tok_s[ml] * H_DIM + (c0 + wn + c * 16 + ncol);
                float v = acc[c][r];
                out[idx] = accumulate ? (out[idx] + v) : v;
            }
        }
    }
}

// ---------------------------------------------------------------------------
extern "C" void kernel_launch(void* const* d_in, const int* in_sizes, int n_in,
                              void* d_out, int out_size, void* d_ws, size_t ws_size,
                              hipStream_t stream) {
    const float* x   = (const float*)d_in[0];   // [1,2048,2048]
    const float* gw  = (const float*)d_in[1];   // [2048,8]
    const float* bgu = (const float*)d_in[2];   // [2048,8192]
    const float* bdw = (const float*)d_in[3];   // [4096,2048]
    const float* egu = (const float*)d_in[4];   // [8,2048,8192]
    const float* edw = (const float*)d_in[5];   // [8,4096,2048]
    float* out = (float*)d_out;

    char* ws = (char*)d_ws;
    float* comb = (float*)ws;                                    // 64 KB
    int* counts = (int*)(ws + 64 * 1024);                        // (E+1) ints
    int* lists  = (int*)(ws + 64 * 1024 + 256);                  // 64 KB
    unsigned short* xb = (unsigned short*)(ws + 64 * 1024 + 256 + 64 * 1024);
    unsigned short* tb = xb + (size_t)NTOK * H_DIM;              // 16 MB bf16

    init_counts_kernel<<<1, 32, 0, stream>>>(counts);
    cvt_x_kernel<<<(NTOK * H_DIM) / 4 / 256, 256, 0, stream>>>(x, xb, (NTOK * H_DIM) / 4);
    router_kernel<<<NTOK / 8, 256, 0, stream>>>(x, gw, comb, counts, lists);

    dim3 ggu(I_DIM / CT, NTOK / MT);   // (32, 32)
    dim3 gdn(H_DIM / CT, NTOK / MT);   // (16, 32)

    // dense base MLP: writes out
    gateup_kernel<<<ggu, 256, 0, stream>>>(xb, bgu, tb, nullptr, nullptr, counts + E_NUM);
    down_kernel<<<gdn, 256, 0, stream>>>(tb, bdw, out, nullptr, counts + E_NUM, 0);

    // experts: gathered rows, comb folded into intermediate, out += ...
    for (int e = 0; e < E_NUM; ++e) {
        gateup_kernel<<<ggu, 256, 0, stream>>>(
            xb, egu + (size_t)e * H_DIM * 2 * I_DIM, tb,
            lists + e * NTOK, comb + e, counts + e);
        down_kernel<<<gdn, 256, 0, stream>>>(
            tb, edw + (size_t)e * I_DIM * H_DIM, out,
            lists + e * NTOK, counts + e, 1);
    }
}
